// RNN_7619271983663
// MI455X (gfx1250) — matrix operand order repaired and checked by reading, not compiled, not run on hardware
//
#include <hip/hip_runtime.h>

// ---------------------------------------------------------------------------
// Problem constants (from reference)
// ---------------------------------------------------------------------------
#define BATCH 64
#define TLEN  2048
#define EDIM  256
#define HDIM  256

typedef __attribute__((ext_vector_type(16))) __bf16 v16bf;
typedef __attribute__((ext_vector_type(8)))  __bf16 v8bf;
typedef __attribute__((ext_vector_type(8)))  float  v8f;

union BF16x16 { v16bf v; v8bf h[2]; };

// Hardware tanh (V_TANH_F32) when the toolchain exposes it.
#if __has_builtin(__builtin_amdgcn_tanhf)
__device__ __forceinline__ float fast_tanh(float x) { return __builtin_amdgcn_tanhf(x); }
#else
__device__ __forceinline__ float fast_tanh(float x) { return tanhf(x); }
#endif

// ---------------------------------------------------------------------------
// Kernel 0: transpose + convert Wxh and Whh to bf16 [N][K] so that WMMA
// B-fragments become two contiguous 16B loads per lane.
// blocks 0..255 -> Wxh, blocks 256..511 -> Whh. One output row per block.
// ---------------------------------------------------------------------------
__global__ __launch_bounds__(256)
void prep_weights(const float* __restrict__ Wxh, const float* __restrict__ Whh,
                  __bf16* __restrict__ WxhT, __bf16* __restrict__ WhhT) {
    const int n = blockIdx.x & 255;
    const float* src = (blockIdx.x < 256) ? Wxh : Whh;
    __bf16*      dst = (blockIdx.x < 256) ? WxhT : WhhT;
    const int k = threadIdx.x;
    dst[n * HDIM + k] = (__bf16)src[k * HDIM + n];
}

// ---------------------------------------------------------------------------
// Kernel 1: fused embedding gather + xproj GEMM (bandwidth-bound, ~6us floor).
// One block = 16 rows of (b*T+t) x all 256 H columns. 256 threads = 8 waves,
// each wave owns 2 column tiles (16x16), K=256 in 8 chunks of 32 (bf16 WMMA).
// Gathered embedding rows staged once in LDS as bf16; Wxh^T fragments come
// from L2 (128 KB, fully resident). xproj stored bf16 (halves HBM traffic).
// ---------------------------------------------------------------------------
__global__ __launch_bounds__(256)
void embed_xproj(const int* __restrict__ x, const float* __restrict__ emb,
                 const __bf16* __restrict__ WxhT, const float* __restrict__ bh,
                 __bf16* __restrict__ xproj) {
    __shared__ __align__(16) __bf16 embS[16 * EDIM];   // 8 KB, row-major [16][256]

    const int  tid     = threadIdx.x;
    const long rowBase = (long)blockIdx.x * 16;

    // stage 16 embedding rows (f32 -> bf16)
    {
        const int  r   = tid >> 4;          // 0..15
        const int  c0  = (tid & 15) * 16;   // 0..240
        const int  tok = x[rowBase + r];
        const float* src = emb + (long)tok * EDIM + c0;
        #pragma unroll
        for (int j = 0; j < 16; ++j) embS[r * EDIM + c0 + j] = (__bf16)src[j];
    }
    __syncthreads();

    const int lane   = tid & 31;
    const int wave   = tid >> 5;     // 0..7
    const int laneR  = lane & 15;    // M (A) / N (B) index within tile
    const int hiHalf = lane >> 4;    // which K half this lane holds
    const int ct0    = wave * 2;     // first of this wave's two column tiles

    v8f acc0 = {}; v8f acc1 = {};
    #pragma unroll
    for (int kc = 0; kc < 8; ++kc) {
        // A fragment: lane<16 -> K {0..7,16..23}; lane>=16 -> K {8..15,24..31}
        BF16x16 A;
        const __bf16* ap = embS + laneR * EDIM + kc * 32 + hiHalf * 8;
        A.h[0] = *(const v8bf*)(ap);
        A.h[1] = *(const v8bf*)(ap + 16);
        // B fragments: lane holds column n, K {0..7,16..23} / {8..15,24..31} from WxhT[n][.]
        BF16x16 B0, B1;
        const __bf16* bp0 = WxhT + (ct0 * 16 + laneR) * HDIM + kc * 32 + hiHalf * 8;
        B0.h[0] = *(const v8bf*)(bp0);
        B0.h[1] = *(const v8bf*)(bp0 + 16);
        const __bf16* bp1 = bp0 + 16 * HDIM;
        B1.h[0] = *(const v8bf*)(bp1);
        B1.h[1] = *(const v8bf*)(bp1 + 16);

        acc0 = __builtin_amdgcn_wmma_f32_16x16x32_bf16(false, A.v, false, B0.v,
                                                       (short)0, acc0, false, false);
        acc1 = __builtin_amdgcn_wmma_f32_16x16x32_bf16(false, A.v, false, B1.v,
                                                       (short)0, acc1, false, false);
    }

    const float bh0 = bh[ct0 * 16 + laneR];
    const float bh1 = bh[ct0 * 16 + 16 + laneR];
    #pragma unroll
    for (int i = 0; i < 8; ++i) {
        // C layout: VGPR i -> row M=i (lanes 0-15) / M=i+8 (lanes 16-31)
        const long row = rowBase + i + hiHalf * 8;
        xproj[row * HDIM + ct0 * 16 + laneR]      = (__bf16)(acc0[i] + bh0);
        xproj[row * HDIM + ct0 * 16 + 16 + laneR] = (__bf16)(acc1[i] + bh1);
    }
}

// ---------------------------------------------------------------------------
// Kernel 2: sequential tanh recurrence + output head (the critical path).
// Single workgroup (1024 threads = 32 waves) pinned to one WGP.
//   h      : LDS ping-pong, bf16 2x[64][256] (64 KB) -> ONE barrier per step
//   Whh^T  : register-resident bf16 B-fragments (2 tiles x 8 k-chunks/wave)
// Per step per wave: prefetched xproj tile, 16x v_wmma_f32_16x16x32_bf16,
// 16x V_TANH_F32, bf16 stores into the other buffer, one barrier.
// Next step's xproj is prefetched as raw bf16 regs during tanh/stores/barrier.
// ---------------------------------------------------------------------------
__global__ __launch_bounds__(1024)
void rnn_scan(const __bf16* __restrict__ xproj, const __bf16* __restrict__ WhhT,
              const float* __restrict__ Why, const float* __restrict__ by,
              float* __restrict__ out) {
    __shared__ __align__(16) __bf16 hS[2 * BATCH * HDIM];   // 64 KB ping-pong

    const int tid    = threadIdx.x;
    const int lane   = tid & 31;
    const int wave   = tid >> 5;     // 0..31
    const int rt     = wave >> 3;    // row (batch) tile 0..3
    const int cp     = wave & 7;     // column-pair index -> col tiles 2cp,2cp+1
    const int laneR  = lane & 15;
    const int hiHalf = lane >> 4;

    // preload Whh B-fragments into registers (128 VGPRs)
    v16bf Bf[2][8];
    #pragma unroll
    for (int j = 0; j < 2; ++j) {
        const int ct = cp * 2 + j;
        #pragma unroll
        for (int kc = 0; kc < 8; ++kc) {
            BF16x16 b;
            const __bf16* bp = WhhT + (ct * 16 + laneR) * HDIM + kc * 32 + hiHalf * 8;
            b.h[0] = *(const v8bf*)(bp);
            b.h[1] = *(const v8bf*)(bp + 16);
            Bf[j][kc] = b.v;
        }
    }

    // h0 = 0 (buffer 0 only; buffer 1 is written before first read)
    #pragma unroll
    for (int i = 0; i < 8; ++i) ((unsigned int*)hS)[tid + i * 1024] = 0u;
    __syncthreads();

    const int col0 = cp * 32 + laneR;      // column of acc0; acc1 is col0+16
    const int rowA = rt * 16 + laneR;      // h row this lane supplies to A

    __bf16* hr = hS;                       // read buffer
    __bf16* hw = hS + BATCH * HDIM;        // write buffer

    // prefetch xproj tile for t=0 (raw bf16, converted at use)
    __bf16 nxr0[8], nxr1[8];
    #pragma unroll
    for (int i = 0; i < 8; ++i) {
        const size_t base = (size_t)(rt * 16 + i + hiHalf * 8) * ((size_t)TLEN * HDIM);
        nxr0[i] = xproj[base + col0];
        nxr1[i] = xproj[base + col0 + 16];
    }

    #pragma unroll 2
    for (int t = 0; t < TLEN; ++t) {
        // acc = xproj[:, t, :] tile (C layout); cvt lands after prior barrier
        v8f acc0, acc1;
        #pragma unroll
        for (int i = 0; i < 8; ++i) { acc0[i] = (float)nxr0[i]; acc1[i] = (float)nxr1[i]; }

        // acc += h @ Whh   (reads hr only)
        #pragma unroll
        for (int kc = 0; kc < 8; ++kc) {
            BF16x16 A;
            const __bf16* ap = hr + rowA * HDIM + kc * 32 + hiHalf * 8;
            A.h[0] = *(const v8bf*)(ap);
            A.h[1] = *(const v8bf*)(ap + 16);
            acc0 = __builtin_amdgcn_wmma_f32_16x16x32_bf16(false, A.v, false, Bf[0][kc],
                                                           (short)0, acc0, false, false);
            acc1 = __builtin_amdgcn_wmma_f32_16x16x32_bf16(false, A.v, false, Bf[1][kc],
                                                           (short)0, acc1, false, false);
        }

        // prefetch next step's xproj (wraps to t=0 at the end: always valid)
        const int tn = (t + 1) & (TLEN - 1);
        #pragma unroll
        for (int i = 0; i < 8; ++i) {
            const size_t base = (size_t)(rt * 16 + i + hiHalf * 8) * ((size_t)TLEN * HDIM)
                              + (size_t)tn * HDIM;
            nxr0[i] = xproj[base + col0];
            nxr1[i] = xproj[base + col0 + 16];
        }

        // hardware tanh + bf16 convert while prefetch is in flight
        __bf16 tv0[8], tv1[8];
        #pragma unroll
        for (int i = 0; i < 8; ++i) {
            tv0[i] = (__bf16)fast_tanh(acc0[i]);
            tv1[i] = (__bf16)fast_tanh(acc1[i]);
        }

        // write new h into the OTHER buffer: never races with readers of hr
        #pragma unroll
        for (int i = 0; i < 8; ++i) {
            const int b = rt * 16 + i + hiHalf * 8;
            hw[b * HDIM + col0]      = tv0[i];
            hw[b * HDIM + col0 + 16] = tv1[i];
        }
        __syncthreads();                    // single barrier per step
        __bf16* tmp = hr; hr = hw; hw = tmp;
    }

    // output head: out[b][o] = h[b,:] . Why[:,o] + by[o]
    if (tid < BATCH * 2) {
        const int b = tid >> 1, o = tid & 1;
        float s = by[o];
        for (int k = 0; k < HDIM; ++k)
            s += (float)hr[b * HDIM + k] * Why[k * 2 + o];
        out[b * 2 + o] = s;
    }
}

// ---------------------------------------------------------------------------
// Launch
//   d_in: x[int], emb[f32], Wxh[f32], Whh[f32], bh[f32], Why[f32], by[f32]
//   ws layout: [0, 128K)    WxhT bf16
//              [128K, 256K) WhhT bf16
//              [256K, +64MiB) xproj bf16  [B*T][H]
// ---------------------------------------------------------------------------
extern "C" void kernel_launch(void* const* d_in, const int* in_sizes, int n_in,
                              void* d_out, int out_size, void* d_ws, size_t ws_size,
                              hipStream_t stream) {
    const int*   x    = (const int*)d_in[0];
    const float* emb  = (const float*)d_in[1];
    const float* Wxh  = (const float*)d_in[2];
    const float* Whh  = (const float*)d_in[3];
    const float* bh   = (const float*)d_in[4];
    const float* Why  = (const float*)d_in[5];
    const float* by   = (const float*)d_in[6];
    float*       outp = (float*)d_out;

    __bf16* WxhT  = (__bf16*)d_ws;
    __bf16* WhhT  = WxhT + EDIM * HDIM;
    __bf16* xproj = (__bf16*)((char*)d_ws + 2 * EDIM * HDIM * sizeof(__bf16));

    prep_weights<<<512, 256, 0, stream>>>(Wxh, Whh, WxhT, WhhT);
    embed_xproj<<<(BATCH * TLEN) / 16, 256, 0, stream>>>(x, emb, WxhT, bh, xproj);
    rnn_scan<<<1, 1024, 0, stream>>>(xproj, WhhT, Why, by, outp);
}
